// MultiheadAttention_33122787787077
// MI455X (gfx1250) — compile-verified, matrix operand order checked by reading, not run on hardware
//
#include <hip/hip_runtime.h>
#include <hip/hip_bf16.h>

// ---- MultiheadAttention, T=2048 B=2 E=1024 H=16 head_dim=64, fp32 in/out ----
// Matmuls via v_wmma_f32_16x16x32_bf16; flash-style attention (scores stay in
// LDS); K/V/Q tiles staged with GLOBAL_LOAD_ASYNC_TO_LDS_B128 when available.

typedef __attribute__((ext_vector_type(16))) __bf16 v16bf;
typedef __attribute__((ext_vector_type(2)))  __bf16 bf16x2;
typedef __attribute__((ext_vector_type(8)))  float  v8f;
typedef __attribute__((ext_vector_type(4)))  float  floatx4;
typedef __attribute__((ext_vector_type(2)))  float  floatx2;
typedef int v4i __attribute__((vector_size(16)));

static constexpr int Tlen = 2048;
static constexpr int Bsz  = 2;
static constexpr int Emb  = 1024;
static constexpr int Hn   = 16;
static constexpr int Hd   = 64;
static constexpr int Mrows = Tlen * Bsz;      // 4096 flattened (t,b) rows
static constexpr int BH    = Bsz * Hn;        // 32 head-batches

// ---------------- helpers ----------------

// scalar fp32 -> bf16 bits via hardware fptrunc
__device__ __forceinline__ unsigned short bf16bits(float f) {
  __bf16 h = (__bf16)f;
  return __builtin_bit_cast(unsigned short, h);
}

// two fp32 -> packed bf16x2 in one uint (v_cvt_pk_bf16_f32 class)
__device__ __forceinline__ unsigned pack_bf16(float lo, float hi) {
#if defined(__AMDGCN__) && __has_builtin(__builtin_amdgcn_cvt_pk_bf16_f32)
  auto v = __builtin_amdgcn_cvt_pk_bf16_f32(lo, hi);
  return __builtin_bit_cast(unsigned, v);
#else
  floatx2 t = {lo, hi};
  bf16x2 r = __builtin_convertvector(t, bf16x2);
  return __builtin_bit_cast(unsigned, r);
#endif
}

#if defined(__AMDGCN__) && !__has_builtin(__builtin_amdgcn_cvt_pk_bf16_f32)
#warning "probe: __builtin_amdgcn_cvt_pk_bf16_f32 NOT available; using convertvector"
#endif

// async global->LDS 16B copy (per-lane) + wait, with sync fallback
#if defined(__AMDGCN__) && __has_builtin(__builtin_amdgcn_global_load_async_to_lds_b128) && __has_builtin(__builtin_amdgcn_s_wait_asynccnt)
#define USE_ASYNC_LDS 1
__device__ __forceinline__ void cp_async16(void* lds, const void* g) {
  __builtin_amdgcn_global_load_async_to_lds_b128(
      (__attribute__((address_space(1))) v4i*)g,
      (__attribute__((address_space(3))) v4i*)lds, 0, 0);
}
__device__ __forceinline__ void async_wait0() {
  __builtin_amdgcn_s_wait_asynccnt(0);
}
#else
#define USE_ASYNC_LDS 0
__device__ __forceinline__ void cp_async16(void* lds, const void* g) {
  *(uint4*)lds = *(const uint4*)g;
}
__device__ __forceinline__ void async_wait0() {}
#endif

#if defined(__AMDGCN__) && !USE_ASYNC_LDS
#warning "probe: async global->LDS builtins NOT available; using sync fallback"
#endif

// Load one 16x32 bf16 WMMA fragment (A or B) from LDS, row-major over K.
// p points at &lds[row * ld] (+ k chunk); per ISA 16-bit A-matrix layout:
// lanes 0-15 hold K {0..7,16..23}, lanes 16-31 hold K {8..15,24..31}.
__device__ __forceinline__ v16bf ld_frag(const unsigned short* p, int half) {
  v16bf f;
#pragma unroll
  for (int i = 0; i < 8; ++i) {
    f[i]     = __builtin_bit_cast(__bf16, p[half * 8 + i]);
    f[8 + i] = __builtin_bit_cast(__bf16, p[16 + half * 8 + i]);
  }
  return f;
}

__device__ __forceinline__ v8f wmma_bf16(v16bf a, v16bf b, v8f c) {
  return __builtin_amdgcn_wmma_f32_16x16x32_bf16(
      false, a, false, b, (short)0, c, false, false);
}

// ---------------------------------------------------------------------------
// Kernel 1: fused QKV projection.  Y = (X @ W_p^T + b_p) [* 1/8 for p==0]
// Output: Q,K -> per-head [b*H+h][t][64] bf16; V -> transposed [b*H+h][64][t]
// so the attention kernel can async-copy all tiles without transposing.
// Block tile 64(M) x 64(N), K-step 64; 8 waves x 2 subtiles x 2 k-steps.
// ---------------------------------------------------------------------------
__global__ __launch_bounds__(256) void qkv_proj_kernel(
    const float* __restrict__ q_in, const float* __restrict__ k_in,
    const float* __restrict__ v_in, const float* __restrict__ w,
    const float* __restrict__ bias,
    unsigned short* __restrict__ Qh, unsigned short* __restrict__ Kh,
    unsigned short* __restrict__ Vt) {
  __shared__ __align__(16) unsigned short sA[64 * 64];
  __shared__ __align__(16) unsigned short sB[64 * 64];

  const int tid  = threadIdx.x;
  const int lane = tid & 31, wv = tid >> 5;
  const int half = lane >> 4, ln = lane & 15;
  const int m0 = blockIdx.x * 64, n0 = blockIdx.y * 64, p = blockIdx.z;

  const float* X = (p == 0) ? q_in : (p == 1) ? k_in : v_in;
  unsigned short* D = (p == 0) ? Qh : (p == 1) ? Kh : Vt;

  const int ni = wv & 3, mi0 = wv >> 2, mi1 = mi0 + 2;
  v8f acc0 = {}, acc1 = {};

  for (int k0 = 0; k0 < Emb; k0 += 64) {
#pragma unroll
    for (int j = 0; j < 2; ++j) {
      int idx = tid + 256 * j;             // 0..511
      int r = idx >> 3, c = (idx & 7) * 8; // 8 fp32 per (thread,pass)
      const floatx4* sa = (const floatx4*)(X + (size_t)(m0 + r) * Emb + k0 + c);
      floatx4 x0 = sa[0], x1 = sa[1];
      const floatx4* sb =
          (const floatx4*)(w + (size_t)(p * Emb + n0 + r) * Emb + k0 + c);
      floatx4 y0 = sb[0], y1 = sb[1];
      unsigned pa[4] = {pack_bf16(x0[0], x0[1]), pack_bf16(x0[2], x0[3]),
                        pack_bf16(x1[0], x1[1]), pack_bf16(x1[2], x1[3])};
      unsigned pb[4] = {pack_bf16(y0[0], y0[1]), pack_bf16(y0[2], y0[3]),
                        pack_bf16(y1[0], y1[1]), pack_bf16(y1[2], y1[3])};
      *(uint4*)&sA[r * 64 + c] = *(uint4*)pa;
      *(uint4*)&sB[r * 64 + c] = *(uint4*)pb;
      if (j == 0 && k0 + 64 < Emb) {
        __builtin_prefetch(X + (size_t)(m0 + r) * Emb + k0 + 64 + c, 0, 1);
        __builtin_prefetch(w + (size_t)(p * Emb + n0 + r) * Emb + k0 + 64 + c, 0, 1);
      }
    }
    __syncthreads();
#pragma unroll
    for (int kk = 0; kk < 64; kk += 32) {
      v16bf bfg = ld_frag(&sB[(ni * 16 + ln) * 64 + kk], half);
      v16bf a0  = ld_frag(&sA[(mi0 * 16 + ln) * 64 + kk], half);
      v16bf a1  = ld_frag(&sA[(mi1 * 16 + ln) * 64 + kk], half);
      acc0 = wmma_bf16(a0, bfg, acc0);
      acc1 = wmma_bf16(a1, bfg, acc1);
    }
    __syncthreads();
  }

  const float scaling = (p == 0) ? 0.125f : 1.0f;  // head_dim^-0.5
  const int nn = n0 + ni * 16 + ln;
  const float bs = bias[p * Emb + nn];
  const int h = nn >> 6, d = nn & 63;
#pragma unroll
  for (int v = 0; v < 8; ++v) {
    int ma = m0 + mi0 * 16 + v + 8 * half;
    int mb = m0 + mi1 * 16 + v + 8 * half;
    float va = (acc0[v] + bs) * scaling;
    float vb = (acc1[v] + bs) * scaling;
    int ta = ma >> 1, ba = ma & 1;
    int tb = mb >> 1, bb = mb & 1;
    if (p == 2) {  // V stored transposed per head: [bh][d][t]
      D[(((size_t)(ba * Hn + h)) * Hd + d) * Tlen + ta] = bf16bits(va);
      D[(((size_t)(bb * Hn + h)) * Hd + d) * Tlen + tb] = bf16bits(vb);
    } else {       // Q,K: [bh][t][d]
      D[(((size_t)(ba * Hn + h)) * Tlen + ta) * Hd + d] = bf16bits(va);
      D[(((size_t)(bb * Hn + h)) * Tlen + tb) * Hd + d] = bf16bits(vb);
    }
  }
}

// ---------------------------------------------------------------------------
// Kernel 2: flash attention per (head-batch, 64-row Q tile).
// Tiles staged via async global->LDS copies; online softmax with 4 threads
// per score row; P@V accumulated in fp32.
// ---------------------------------------------------------------------------
__global__ __launch_bounds__(256) void attn_kernel(
    const unsigned short* __restrict__ Qh, const unsigned short* __restrict__ Kh,
    const unsigned short* __restrict__ Vt, unsigned short* __restrict__ Ctx) {
  __shared__ __align__(16) unsigned short sQ[64 * 64];   // [tq][d]
  __shared__ __align__(16) unsigned short sK[64 * 64];   // [s][d]
  __shared__ __align__(16) unsigned short sVt[64 * 64];  // [d][s]
  __shared__ __align__(16) float sS[64 * 64];            // scores fp32
  __shared__ __align__(16) unsigned short sP[64 * 64];   // exp(scores) bf16
  __shared__ float sRedMax[4 * 64];
  __shared__ float sRedSum[4 * 64];
  __shared__ float sAlpha[64];
  __shared__ float sInvL[64];

  const int tid  = threadIdx.x;
  const int lane = tid & 31, wv = tid >> 5;
  const int half = lane >> 4, ln = lane & 15;
  const int rrow = tid & 63;   // softmax row owned (with 3 partners)
  const int quad = tid >> 6;   // which 16-column slice
  const int t0 = blockIdx.x * 64;
  const int bh = blockIdx.y;
  const size_t hb = (size_t)bh * Tlen * Hd;   // same elem count for Q/K/Vt

  // stage Q tile once (async)
#pragma unroll
  for (int j = 0; j < 2; ++j) {
    int idx = tid + 256 * j;
    int r = idx >> 3, c = (idx & 7) * 8;
    cp_async16(&sQ[r * 64 + c], Qh + hb + (size_t)(t0 + r) * Hd + c);
  }

  const int ni = wv & 3, mi0 = wv >> 2, mi1 = mi0 + 2;
  v8f o0 = {}, o1 = {};
  float row_m = -1e30f, row_l = 0.0f;

  for (int s0 = 0; s0 < Tlen; s0 += 64) {
    // stage K [s][d] and pre-transposed V [d][s] (async b128)
#pragma unroll
    for (int j = 0; j < 2; ++j) {
      int idx = tid + 256 * j;
      int r = idx >> 3, c = (idx & 7) * 8;
      cp_async16(&sK[r * 64 + c], Kh + hb + (size_t)(s0 + r) * Hd + c);
      cp_async16(&sVt[r * 64 + c], Vt + hb + (size_t)r * Tlen + s0 + c);
    }
    async_wait0();
    __syncthreads();

    // scores = Q @ K^T  (both K-major over d)
    v8f sc0 = {}, sc1 = {};
#pragma unroll
    for (int kk = 0; kk < Hd; kk += 32) {
      v16bf bk = ld_frag(&sK[(ni * 16 + ln) * 64 + kk], half);
      v16bf a0 = ld_frag(&sQ[(mi0 * 16 + ln) * 64 + kk], half);
      v16bf a1 = ld_frag(&sQ[(mi1 * 16 + ln) * 64 + kk], half);
      sc0 = wmma_bf16(a0, bk, sc0);
      sc1 = wmma_bf16(a1, bk, sc1);
    }
#pragma unroll
    for (int v = 0; v < 8; ++v) {
      sS[(mi0 * 16 + v + 8 * half) * 64 + ni * 16 + ln] = sc0[v];
      sS[(mi1 * 16 + v + 8 * half) * 64 + ni * 16 + ln] = sc1[v];
    }
    __syncthreads();

    // online softmax: 4 threads per row, 16 columns each
    const float* srow = &sS[rrow * 64 + quad * 16];
    float mc = -1e30f;
#pragma unroll
    for (int j = 0; j < 16; ++j) mc = fmaxf(mc, srow[j]);
    sRedMax[quad * 64 + rrow] = mc;
    __syncthreads();

    float mnew = row_m;
#pragma unroll
    for (int q = 0; q < 4; ++q) mnew = fmaxf(mnew, sRedMax[q * 64 + rrow]);
    float alpha = __expf(row_m - mnew);
    float lsum = 0.0f;
#pragma unroll
    for (int j = 0; j < 16; j += 2) {
      float e0 = __expf(srow[j] - mnew);
      float e1 = __expf(srow[j + 1] - mnew);
      *(unsigned*)&sP[rrow * 64 + quad * 16 + j] = pack_bf16(e0, e1);
      lsum += e0 + e1;
    }
    sRedSum[quad * 64 + rrow] = lsum;
    if (quad == 0) sAlpha[rrow] = alpha;
    __syncthreads();

    float ltot = 0.0f;
#pragma unroll
    for (int q = 0; q < 4; ++q) ltot += sRedSum[q * 64 + rrow];
    row_l = row_l * alpha + ltot;
    row_m = mnew;

    // rescale accumulators, then O += P @ V
#pragma unroll
    for (int v = 0; v < 8; ++v) {
      o0[v] *= sAlpha[mi0 * 16 + v + 8 * half];
      o1[v] *= sAlpha[mi1 * 16 + v + 8 * half];
    }
#pragma unroll
    for (int kk = 0; kk < 64; kk += 32) {
      v16bf bv = ld_frag(&sVt[(ni * 16 + ln) * 64 + kk], half);
      v16bf p0 = ld_frag(&sP[(mi0 * 16 + ln) * 64 + kk], half);
      v16bf p1 = ld_frag(&sP[(mi1 * 16 + ln) * 64 + kk], half);
      o0 = wmma_bf16(p0, bv, o0);
      o1 = wmma_bf16(p1, bv, o1);
    }
    __syncthreads();
  }

  if (tid < 64) sInvL[tid] = (row_l > 0.0f) ? (1.0f / row_l) : 0.0f;
  __syncthreads();

  // write context: [row m = t*B + b][h*64 + d] bf16
  const int b = bh >> 4, h = bh & 15;
  const int d = ni * 16 + ln;
#pragma unroll
  for (int v = 0; v < 8; ++v) {
    int ra = mi0 * 16 + v + 8 * half;
    int rb = mi1 * 16 + v + 8 * half;
    float va = o0[v] * sInvL[ra];
    float vb = o1[v] * sInvL[rb];
    Ctx[((size_t)((t0 + ra) * Bsz + b)) * Emb + h * Hd + d] = bf16bits(va);
    Ctx[((size_t)((t0 + rb) * Bsz + b)) * Emb + h * Hd + d] = bf16bits(vb);
  }
}

// ---------------------------------------------------------------------------
// Kernel 3: output projection.  out = Ctx @ Wout^T + bout  (fp32 out)
// Ctx tile staged async (already bf16); W tile converted while staging.
// ---------------------------------------------------------------------------
__global__ __launch_bounds__(256) void out_proj_kernel(
    const unsigned short* __restrict__ Ctx, const float* __restrict__ w,
    const float* __restrict__ bias, float* __restrict__ out) {
  __shared__ __align__(16) unsigned short sA[64 * 64];
  __shared__ __align__(16) unsigned short sB[64 * 64];

  const int tid  = threadIdx.x;
  const int lane = tid & 31, wv = tid >> 5;
  const int half = lane >> 4, ln = lane & 15;
  const int m0 = blockIdx.x * 64, n0 = blockIdx.y * 64;
  const int ni = wv & 3, mi0 = wv >> 2, mi1 = mi0 + 2;
  v8f acc0 = {}, acc1 = {};

  for (int k0 = 0; k0 < Emb; k0 += 64) {
#pragma unroll
    for (int j = 0; j < 2; ++j) {
      int idx = tid + 256 * j;
      int r = idx >> 3, c = (idx & 7) * 8;
      cp_async16(&sA[r * 64 + c], Ctx + (size_t)(m0 + r) * Emb + k0 + c);
      const floatx4* sbp =
          (const floatx4*)(w + (size_t)(n0 + r) * Emb + k0 + c);
      floatx4 y0 = sbp[0], y1 = sbp[1];
      unsigned pb[4] = {pack_bf16(y0[0], y0[1]), pack_bf16(y0[2], y0[3]),
                        pack_bf16(y1[0], y1[1]), pack_bf16(y1[2], y1[3])};
      *(uint4*)&sB[r * 64 + c] = *(uint4*)pb;
      if (j == 0 && k0 + 64 < Emb) {
        __builtin_prefetch(w + (size_t)(n0 + r) * Emb + k0 + 64 + c, 0, 1);
      }
    }
    async_wait0();
    __syncthreads();
#pragma unroll
    for (int kk = 0; kk < 64; kk += 32) {
      v16bf bfg = ld_frag(&sB[(ni * 16 + ln) * 64 + kk], half);
      v16bf a0  = ld_frag(&sA[(mi0 * 16 + ln) * 64 + kk], half);
      v16bf a1  = ld_frag(&sA[(mi1 * 16 + ln) * 64 + kk], half);
      acc0 = wmma_bf16(a0, bfg, acc0);
      acc1 = wmma_bf16(a1, bfg, acc1);
    }
    __syncthreads();
  }

  const int nn = n0 + ni * 16 + ln;
  const float bs = bias[nn];
#pragma unroll
  for (int v = 0; v < 8; ++v) {
    int ma = m0 + mi0 * 16 + v + 8 * half;
    int mb = m0 + mi1 * 16 + v + 8 * half;
    out[(size_t)ma * Emb + nn] = acc0[v] + bs;
    out[(size_t)mb * Emb + nn] = acc1[v] + bs;
  }
}

// ---------------------------------------------------------------------------
extern "C" void kernel_launch(void* const* d_in, const int* in_sizes, int n_in,
                              void* d_out, int out_size, void* d_ws, size_t ws_size,
                              hipStream_t stream) {
  const float* q    = (const float*)d_in[0];
  const float* k    = (const float*)d_in[1];
  const float* v    = (const float*)d_in[2];
  const float* wqkv = (const float*)d_in[3];
  const float* bqkv = (const float*)d_in[4];
  const float* wout = (const float*)d_in[5];
  const float* bout = (const float*)d_in[6];
  float* out = (float*)d_out;

  const size_t headElems = (size_t)BH * Tlen * Hd;  // 4M bf16 = 8 MB each
  unsigned short* Qh  = (unsigned short*)d_ws;
  unsigned short* Kh  = Qh + headElems;
  unsigned short* Vt  = Kh + headElems;             // V transposed per head
  unsigned short* Ctx = Vt + headElems;             // 4096*1024 bf16 = 8 MB

  qkv_proj_kernel<<<dim3(Mrows / 64, Emb / 64, 3), 256, 0, stream>>>(
      q, k, v, wqkv, bqkv, Qh, Kh, Vt);

  attn_kernel<<<dim3(Tlen / 64, BH), 256, 0, stream>>>(Qh, Kh, Vt, Ctx);

  out_proj_kernel<<<dim3(Mrows / 64, Emb / 64), 256, 0, stream>>>(
      Ctx, wout, bout, out);
}
